// SSIMLoss_53850299957381
// MI455X (gfx1250) — matrix operand order repaired and checked by reading, not compiled, not run on hardware
//
#include <hip/hip_runtime.h>

typedef __attribute__((ext_vector_type(16))) _Float16 v16h;
typedef __attribute__((ext_vector_type(8)))  float    v8f;
typedef __attribute__((ext_vector_type(4)))  float    v4f;

#define IMG   512
#define TILE  16
#define WPB   4            // waves per block (128 threads, wave32)
#define NTHR  (WPB * 32)
#define C1    1.0e-4f
#define C2    9.0e-4f

#if defined(__gfx1250__)
#  define USE_ASYNC 1
#else
#  define USE_ASYNC 0
#endif

// Normalized 11-tap Gaussian, sigma = 1.5 (matches reference _build_window)
__device__ __constant__ float GW[11] = {
    0.00102838f, 0.00759880f, 0.03600077f, 0.10936060f, 0.21300553f,
    0.26601222f, 0.21300553f, 0.10936060f, 0.03600077f, 0.00759880f,
    0.00102838f};

// Branchless banded-weight lookup: clamp -> unconditional load -> select.
__device__ __forceinline__ float gwc(int j) {
    int jc = j < 0 ? 0 : (j > 10 ? 10 : j);
    float v = GW[jc];
    return (j == jc) ? v : 0.0f;
}

// One 16x16 output tile per wave. Separable 11-tap conv as band-matrix WMMAs:
// H = F(26x32) x Wh(32x16)  then  V = Wv(16x32) x H(32x16).
__global__ __launch_bounds__(NTHR) void ssim_tiles(const float* __restrict__ pred,
                                                   const float* __restrict__ targ,
                                                   float* __restrict__ partial) {
    // Per-wave input staging: 2 images x 32 rows x 40 cols f32 (40 KB total)
    __shared__ float    Stage[WPB][2][32][40];
    // Per-wave H staging: 5 fields x 32 rows x 16 cols f16 (20 KB total)
    __shared__ _Float16 Hl[WPB][5][32][16];
    __shared__ float    red[NTHR];

    const int tid   = threadIdx.x;
    const int w     = tid >> 5;
    const int lane  = tid & 31;
    const int half  = lane >> 4;
    const int ln    = lane & 15;           // M row (A) / N col (B,C,D)
    const int kb    = half << 4;           // unused (B operands use abase too)
    const int abase = half << 3;           // A-operand K base

    const int tileId = blockIdx.x * WPB + w;
    const int plane  = tileId >> 10;            // 32x32 tiles per 512x512 plane
    const int tip    = tileId & 1023;
    const int tr     = (tip >> 5) << 4;
    const int tc     = (tip & 31) << 4;
    const size_t pbase = (size_t)plane * (size_t)(IMG * IMG);

    // ---- Stage 32x40 halo tiles (clamped addresses; masked at consume) ----
    // Staged LDS col g*4+j  <->  image col tc-8+4g+j ; row s <-> tr-5+s.
#pragma unroll
    for (int inst = 0; inst < 10; ++inst) {
        int gid = inst * 32 + lane;              // 0..319 : 32 rows x 10 groups
        int s   = gid / 10;
        int g   = gid - s * 10;
        int ir  = tr - 5 + s;        ir = ir < 0 ? 0 : (ir > IMG - 1 ? IMG - 1 : ir);
        int ic  = tc - 8 + (g << 2); ic = ic < 0 ? 0 : (ic > IMG - 4 ? IMG - 4 : ic);
        const float* gp = pred + pbase + (size_t)ir * IMG + ic;
        const float* gt = targ + pbase + (size_t)ir * IMG + ic;
        float* lp = &Stage[w][0][s][g << 2];
        float* lt = &Stage[w][1][s][g << 2];
#if USE_ASYNC
        // VDST = LDS byte address (low 32 bits of generic shared pointer),
        // VADDR = 64-bit global address, SADDR = off. Tracked by ASYNCcnt.
        unsigned lpo = (unsigned)(unsigned long long)lp;
        unsigned lto = (unsigned)(unsigned long long)lt;
        asm volatile("global_load_async_to_lds_b128 %0, %1, off"
                     :: "v"(lpo), "v"(gp) : "memory");
        asm volatile("global_load_async_to_lds_b128 %0, %1, off"
                     :: "v"(lto), "v"(gt) : "memory");
#else
        *(v4f*)lp = *(const v4f*)gp;
        *(v4f*)lt = *(const v4f*)gt;
#endif
    }
#if USE_ASYNC
    asm volatile("s_wait_asynccnt 0x0" ::: "memory");
#endif
    __syncthreads();

    // ---- Band-matrix weight operands (same for all fields/chunks) ----
    v16h bh, av;
#pragma unroll
    for (int i = 0; i < 16; ++i) {
        bh[i] = (_Float16)gwc(((i < 8) ? (abase + i) : (16 + abase + (i - 8))) - ln); // Wh[k][n]=g[k-n]
        int ka = (i < 8) ? (abase + i) : (16 + abase + (i - 8));  // A K-index map
        av[i] = (_Float16)gwc(ka - ln);                           // Wv[m][k]=g[k-m]
    }

    const v8f cz = {};

    // ---------------- Phase 1: horizontal pass (10 WMMAs) ----------------
#pragma unroll
    for (int ch = 0; ch < 2; ++ch) {
        const int s    = (ch << 4) + ln;                 // A row M = ln
        const int row  = tr - 5 + s;
        const bool rowOk = (unsigned)row < (unsigned)IMG;

        v16h ax, ay, axx, ayy, axy;
#pragma unroll
        for (int i = 0; i < 16; ++i) {
            int k   = (i < 8) ? (abase + i) : (16 + abase + (i - 8));
            int col = tc - 5 + k;
            bool ok = rowOk && ((unsigned)col < (unsigned)IMG);
            float pv = Stage[w][0][s][k + 3];
            float tv = Stage[w][1][s][k + 3];
            pv = ok ? pv : 0.0f;
            tv = ok ? tv : 0.0f;
            ax[i]  = (_Float16)pv;
            ay[i]  = (_Float16)tv;
            axx[i] = (_Float16)(pv * pv);
            ayy[i] = (_Float16)(tv * tv);
            axy[i] = (_Float16)(pv * tv);
        }

        v8f h;
        h = __builtin_amdgcn_wmma_f32_16x16x32_f16(false, ax, false, bh, (short)0, cz, false, false);
#pragma unroll
        for (int r = 0; r < 8; ++r) Hl[w][0][(ch << 4) + r + (half << 3)][ln] = (_Float16)h[r];
        h = __builtin_amdgcn_wmma_f32_16x16x32_f16(false, ay, false, bh, (short)0, cz, false, false);
#pragma unroll
        for (int r = 0; r < 8; ++r) Hl[w][1][(ch << 4) + r + (half << 3)][ln] = (_Float16)h[r];
        h = __builtin_amdgcn_wmma_f32_16x16x32_f16(false, axx, false, bh, (short)0, cz, false, false);
#pragma unroll
        for (int r = 0; r < 8; ++r) Hl[w][2][(ch << 4) + r + (half << 3)][ln] = (_Float16)h[r];
        h = __builtin_amdgcn_wmma_f32_16x16x32_f16(false, ayy, false, bh, (short)0, cz, false, false);
#pragma unroll
        for (int r = 0; r < 8; ++r) Hl[w][3][(ch << 4) + r + (half << 3)][ln] = (_Float16)h[r];
        h = __builtin_amdgcn_wmma_f32_16x16x32_f16(false, axy, false, bh, (short)0, cz, false, false);
#pragma unroll
        for (int r = 0; r < 8; ++r) Hl[w][4][(ch << 4) + r + (half << 3)][ln] = (_Float16)h[r];
    }

    __syncthreads();

    // ---------------- Phase 2: vertical pass (5 WMMAs) ----------------
    v16h b0, b1, b2, b3, b4;
#pragma unroll
    for (int i = 0; i < 16; ++i) {
        int kk = (i < 8) ? (abase + i) : (16 + abase + (i - 8));
        b0[i] = Hl[w][0][kk][ln];
        b1[i] = Hl[w][1][kk][ln];
        b2[i] = Hl[w][2][kk][ln];
        b3[i] = Hl[w][3][kk][ln];
        b4[i] = Hl[w][4][kk][ln];
    }
    v8f dMx = __builtin_amdgcn_wmma_f32_16x16x32_f16(false, av, false, b0, (short)0, cz, false, false);
    v8f dMy = __builtin_amdgcn_wmma_f32_16x16x32_f16(false, av, false, b1, (short)0, cz, false, false);
    v8f dXX = __builtin_amdgcn_wmma_f32_16x16x32_f16(false, av, false, b2, (short)0, cz, false, false);
    v8f dYY = __builtin_amdgcn_wmma_f32_16x16x32_f16(false, av, false, b3, (short)0, cz, false, false);
    v8f dXY = __builtin_amdgcn_wmma_f32_16x16x32_f16(false, av, false, b4, (short)0, cz, false, false);

    // ---------------- SSIM map + deterministic block reduction -----------
    float acc = 0.0f;
#pragma unroll
    for (int r = 0; r < 8; ++r) {
        float mx = dMx[r], my = dMy[r];
        float mx2 = mx * mx, my2 = my * my, mxy = mx * my;
        float sx  = dXX[r] - mx2;
        float sy  = dYY[r] - my2;
        float sxy = dXY[r] - mxy;
        float num = (2.0f * mxy + C1) * (2.0f * sxy + C2);
        float den = (mx2 + my2 + C1) * (sx + sy + C2);
        acc += num / den;
    }
    red[tid] = acc;
    __syncthreads();
#pragma unroll
    for (int s = NTHR / 2; s > 0; s >>= 1) {
        if (tid < s) red[tid] += red[tid + s];
        __syncthreads();
    }
    if (tid == 0) partial[blockIdx.x] = red[0];
}

__global__ __launch_bounds__(256) void ssim_final(const float* __restrict__ partial,
                                                  float* __restrict__ out,
                                                  int n, float invN) {
    __shared__ float red[256];
    float s = 0.0f;
    for (int i = threadIdx.x; i < n; i += 256) s += partial[i];  // fixed order
    red[threadIdx.x] = s;
    __syncthreads();
#pragma unroll
    for (int st = 128; st > 0; st >>= 1) {
        if (threadIdx.x < st) red[threadIdx.x] += red[threadIdx.x + st];
        __syncthreads();
    }
    if (threadIdx.x == 0) out[0] = 1.0f - red[0] * invN;
}

extern "C" void kernel_launch(void* const* d_in, const int* in_sizes, int n_in,
                              void* d_out, int out_size, void* d_ws, size_t ws_size,
                              hipStream_t stream) {
    const float* pred = (const float*)d_in[0];
    const float* targ = (const float*)d_in[1];
    float* out = (float*)d_out;
    float* partial = (float*)d_ws;

    const int total   = in_sizes[0];                            // 16*3*512*512
    const int planes  = total / (IMG * IMG);                    // 48
    const int tiles   = planes * (IMG / TILE) * (IMG / TILE);   // 49152
    const int blocks1 = tiles / WPB;                            // 12288

    ssim_tiles<<<blocks1, NTHR, 0, stream>>>(pred, targ, partial);
    ssim_final<<<1, 256, 0, stream>>>(partial, out, blocks1, 1.0f / (float)total);
}
